// ConvLayer_15161234555428
// MI455X (gfx1250) — matrix operand order repaired and checked by reading, not compiled, not run on hardware
//
#include <hip/hip_runtime.h>

#define NN 50000
#define NE 640000

typedef __attribute__((ext_vector_type(4)))  float          f32x4;
typedef __attribute__((ext_vector_type(4)))  unsigned int   u32x4;
typedef __attribute__((ext_vector_type(4)))  unsigned short u16x4;
typedef __attribute__((ext_vector_type(16))) __bf16         v16bf;
typedef __attribute__((ext_vector_type(8)))  float          v8f;

union FragCast { u32x4 u[2]; v16bf v; };

__device__ __forceinline__ unsigned short f32_to_bf16(float f) {
  unsigned int u = __float_as_uint(f);
  u += 0x7fffu + ((u >> 16) & 1u);   // round-to-nearest-even
  return (unsigned short)(u >> 16);
}

// A-fragment for V_WMMA_F32_16X16X32_BF16 from row-major bf16 tile in LDS.
// lane 0-15: row=lane,    k = kt*32 + {0..7, 16..23}
// lane 16-31: row=lane-16, k = kt*32 + {8..15, 24..31}
__device__ __forceinline__ v16bf load_a_frag(const unsigned short* lds, int stride,
                                             int rowBase, int kt, int lane) {
  int row = rowBase + (lane & 15);
  int k0  = kt * 32 + ((lane & 16) ? 8 : 0);
  const unsigned short* p = lds + row * stride + k0;
  FragCast f;
  f.u[0] = *(const u32x4*)(p);        // 8 bf16
  f.u[1] = *(const u32x4*)(p + 16);   // 8 bf16, +16 elements along K
  return f.v;
}

// B-fragment from pre-packed weights: per (nt,kt,lane) 16 contiguous bf16.
__device__ __forceinline__ v16bf load_b_frag(const unsigned short* pk, int nTK,
                                             int nt, int kt, int lane) {
  const unsigned short* p = pk + (size_t)(((nt * nTK) + kt) * 32 + lane) * 16;
  FragCast f;
  f.u[0] = *(const u32x4*)(p);
  f.u[1] = *(const u32x4*)(p + 8);
  return f.v;
}

// One hidden MLP layer (16 rows per wave, 128 output cols) with bias + leakyReLU(0.2),
// result stored as bf16 into LDS (stride 136, 16B-aligned, bank-conflict-free padding).
template <int NTK>
__device__ __forceinline__ void hidden_layer(const unsigned short* inLds, int inStride,
                                             const unsigned short* packW,
                                             const float* __restrict__ bias,
                                             unsigned short* outLds, int rowBase, int lane) {
  for (int nt = 0; nt < 8; ++nt) {
    v8f acc = {0.f, 0.f, 0.f, 0.f, 0.f, 0.f, 0.f, 0.f};
#pragma unroll
    for (int kt = 0; kt < NTK; ++kt) {
      v16bf A = load_a_frag(inLds, inStride, rowBase, kt, lane);
      v16bf B = load_b_frag(packW, NTK, nt, kt, lane);
      acc = __builtin_amdgcn_wmma_f32_16x16x32_bf16(false, A, false, B, (short)0, acc,
                                                    false, false);
    }
    int col = nt * 16 + (lane & 15);
    float b = bias[col];
    int r0 = rowBase + ((lane & 16) ? 8 : 0);
#pragma unroll
    for (int r = 0; r < 8; ++r) {
      float x = acc[r] + b;
      x = (x > 0.f) ? x : 0.2f * x;
      outLds[(r0 + r) * 136 + col] = f32_to_bf16(x);
    }
  }
}

// ---------------------------------------------------------------------------
// Weight packing: f32 row-major [K,128] -> bf16 WMMA-B fragments.
__global__ __launch_bounds__(256) void pack_weights_kernel(
    const float* __restrict__ We1, const float* __restrict__ We2,
    const float* __restrict__ We3, const float* __restrict__ Wv1,
    const float* __restrict__ Wv2, const float* __restrict__ Wv3,
    unsigned short* __restrict__ pw) {
  int t = blockIdx.x * 256 + threadIdx.x;
  const float* src;
  unsigned short* dst;
  int K;
  if (t < 3072)      { src = We1; dst = pw + 0;      K = 384; }
  else if (t < 4096) { t -= 3072; src = We2; dst = pw + 49152;  K = 128; }
  else if (t < 5120) { t -= 4096; src = We3; dst = pw + 65536;  K = 128; }
  else if (t < 7168) { t -= 5120; src = Wv1; dst = pw + 81920;  K = 256; }
  else if (t < 8192) { t -= 7168; src = Wv2; dst = pw + 114688; K = 128; }
  else if (t < 9216) { t -= 8192; src = Wv3; dst = pw + 131072; K = 128; }
  else return;
  int lane = t & 31;
  int ft   = t >> 5;
  int nTK  = K / 32;
  int kt   = ft % nTK;
  int nt   = ft / nTK;
  int col  = nt * 16 + (lane & 15);
  int kb   = kt * 32 + ((lane & 16) ? 8 : 0);
  unsigned short* d = dst + (size_t)(((nt * nTK) + kt) * 32 + lane) * 16;
#pragma unroll
  for (int i = 0; i < 16; ++i) d[i] = f32_to_bf16(src[(size_t)(kb + (i < 8 ? i : i + 8)) * 128 + col]);
}

// ---------------------------------------------------------------------------
// Edge MLP: gather, 3-layer bf16 WMMA MLP, raw output + BN stats + segment scatter.
__global__ __launch_bounds__(256) void edge_mlp_kernel(
    const float* __restrict__ node_fea, const float* __restrict__ edge_fea,
    const int* __restrict__ idx1, const int* __restrict__ idx2,
    const unsigned short* __restrict__ pWe1, const unsigned short* __restrict__ pWe2,
    const unsigned short* __restrict__ pWe3,
    const float* __restrict__ be1, const float* __restrict__ be2,
    const float* __restrict__ be3,
    float* __restrict__ ek_raw, float* __restrict__ seg_sum,
    float* __restrict__ cnt, float* __restrict__ se_sum, float* __restrict__ se_sq) {
  extern __shared__ unsigned char smem[];
  unsigned short* sh_z  = (unsigned short*)smem;             // 128 x 392 bf16 (100352 B)
  unsigned short* sh_h1 = (unsigned short*)(smem + 100352);  // 128 x 136 bf16 (34816 B)
  int* s_i1 = (int*)(smem + 135168);
  int* s_i2 = (int*)(smem + 135680);

  const int tid = threadIdx.x;
  const int e0  = blockIdx.x * 128;

  if (tid < 128) {
    int i1 = idx1[e0 + tid];
    s_i1[tid] = i1;
    atomicAdd(&cnt[i1], 1.0f);
  } else {
    s_i2[tid - 128] = idx2[e0 + (tid - 128)];
  }
  __syncthreads();

  // Gather z = [node[idx1] | node[idx2] | edge] -> bf16 LDS tile (stride 392).
  for (int it = tid; it < 128 * 96; it += 256) {
    int row = it / 96;
    int c   = (it % 96) * 4;
    const float* src;
    if (c < 128)      src = node_fea + (size_t)s_i1[row] * 128 + c;
    else if (c < 256) src = node_fea + (size_t)s_i2[row] * 128 + (c - 128);
    else              src = edge_fea + (size_t)(e0 + row) * 128 + (c - 256);
    f32x4 v = *(const f32x4*)src;
    u16x4 h;
#pragma unroll
    for (int j = 0; j < 4; ++j) h[j] = f32_to_bf16(v[j]);
    *(u16x4*)(sh_z + row * 392 + c) = h;
  }
  __syncthreads();

  const int lane    = tid & 31;
  const int rowBase = (tid >> 5) * 16;

  hidden_layer<12>(sh_z, 392, pWe1, be1, sh_h1, rowBase, lane);   // 384 -> 128
  __syncthreads();                                                // z region reused as h2
  hidden_layer<4>(sh_h1, 136, pWe2, be2, sh_z, rowBase, lane);    // 128 -> 128

  // Layer 3 (no relu): raw output -> global, BN stats, segment scatter.
  for (int nt = 0; nt < 8; ++nt) {
    v8f acc = {0.f, 0.f, 0.f, 0.f, 0.f, 0.f, 0.f, 0.f};
#pragma unroll
    for (int kt = 0; kt < 4; ++kt) {
      v16bf A = load_a_frag(sh_z, 136, rowBase, kt, lane);
      v16bf B = load_b_frag(pWe3, 4, nt, kt, lane);
      acc = __builtin_amdgcn_wmma_f32_16x16x32_bf16(false, A, false, B, (short)0, acc,
                                                    false, false);
    }
    int col = nt * 16 + (lane & 15);
    float b = be3[col];
    int r0  = rowBase + ((lane & 16) ? 8 : 0);
    float vals[8];
    float s = 0.f, ss = 0.f;
#pragma unroll
    for (int r = 0; r < 8; ++r) {
      float x = acc[r] + b;
      vals[r] = x;
      s += x;
      ss += x * x;
    }
    s  += __shfl_xor(s, 16, 32);   // pair lanes holding same column
    ss += __shfl_xor(ss, 16, 32);
    if (lane < 16) {
      atomicAdd(&se_sum[col], s);
      atomicAdd(&se_sq[col], ss);
    }
#pragma unroll
    for (int r = 0; r < 8; ++r) {
      int row = r0 + r;
      ek_raw[(size_t)(e0 + row) * 128 + col] = vals[r];
      atomicAdd(&seg_sum[(size_t)s_i1[row] * 128 + col], vals[r]);
    }
  }
}

// ---------------------------------------------------------------------------
// Node MLP: z2 = [BN-affine segment mean | node_fea], 3-layer WMMA MLP -> raw + stats.
__global__ __launch_bounds__(256) void node_mlp_kernel(
    const float* __restrict__ node_fea,
    const float* __restrict__ seg_sum, const float* __restrict__ cnt,
    const float* __restrict__ a_e, const float* __restrict__ b_e,
    const unsigned short* __restrict__ pWv1, const unsigned short* __restrict__ pWv2,
    const unsigned short* __restrict__ pWv3,
    const float* __restrict__ bv1, const float* __restrict__ bv2,
    const float* __restrict__ bv3,
    float* __restrict__ vi_raw, float* __restrict__ sv_sum, float* __restrict__ sv_sq) {
  extern __shared__ unsigned char smem[];
  unsigned short* sh_z  = (unsigned short*)smem;            // 128 x 264 bf16 (67584 B)
  unsigned short* sh_h1 = (unsigned short*)(smem + 67584);  // 128 x 136 bf16

  const int tid = threadIdx.x;
  const int n0  = blockIdx.x * 128;

  for (int it = tid; it < 128 * 64; it += 256) {
    int row = it >> 6;
    int c   = (it & 63) * 4;
    int n   = n0 + row;
    f32x4 v = {0.f, 0.f, 0.f, 0.f};
    if (n < NN) {
      if (c < 128) {
        // vi_e_bar = (a_e*segsum_raw + cnt*b_e) / max(cnt,1)
        f32x4 s4 = *(const f32x4*)(seg_sum + (size_t)n * 128 + c);
        f32x4 a4 = *(const f32x4*)(a_e + c);
        f32x4 b4 = *(const f32x4*)(b_e + c);
        float cn  = cnt[n];
        float inv = 1.0f / fmaxf(cn, 1.0f);
#pragma unroll
        for (int j = 0; j < 4; ++j) v[j] = (a4[j] * s4[j] + cn * b4[j]) * inv;
      } else {
        v = *(const f32x4*)(node_fea + (size_t)n * 128 + (c - 128));
      }
    }
    u16x4 h;
#pragma unroll
    for (int j = 0; j < 4; ++j) h[j] = f32_to_bf16(v[j]);
    *(u16x4*)(sh_z + row * 264 + c) = h;
  }
  __syncthreads();

  const int lane    = tid & 31;
  const int rowBase = (tid >> 5) * 16;

  hidden_layer<8>(sh_z, 264, pWv1, bv1, sh_h1, rowBase, lane);   // 256 -> 128
  __syncthreads();
  hidden_layer<4>(sh_h1, 136, pWv2, bv2, sh_z, rowBase, lane);   // 128 -> 128

  for (int nt = 0; nt < 8; ++nt) {
    v8f acc = {0.f, 0.f, 0.f, 0.f, 0.f, 0.f, 0.f, 0.f};
#pragma unroll
    for (int kt = 0; kt < 4; ++kt) {
      v16bf A = load_a_frag(sh_z, 136, rowBase, kt, lane);
      v16bf B = load_b_frag(pWv3, 4, nt, kt, lane);
      acc = __builtin_amdgcn_wmma_f32_16x16x32_bf16(false, A, false, B, (short)0, acc,
                                                    false, false);
    }
    int col = nt * 16 + (lane & 15);
    float b = bv3[col];
    int r0  = rowBase + ((lane & 16) ? 8 : 0);
    float s = 0.f, ss = 0.f;
#pragma unroll
    for (int r = 0; r < 8; ++r) {
      int g = n0 + r0 + r;
      float x = acc[r] + b;
      if (g < NN) {
        s += x;
        ss += x * x;
        vi_raw[(size_t)g * 128 + col] = x;
      }
    }
    s  += __shfl_xor(s, 16, 32);
    ss += __shfl_xor(ss, 16, 32);
    if (lane < 16) {
      atomicAdd(&sv_sum[col], s);
      atomicAdd(&sv_sq[col], ss);
    }
  }
}

// ---------------------------------------------------------------------------
// BN(train) stats -> per-column affine a,b:  BN(x) = a*x + b
__global__ void finalize_stats_kernel(const float* __restrict__ sum,
                                      const float* __restrict__ sq,
                                      const float* __restrict__ gamma,
                                      const float* __restrict__ beta,
                                      float* __restrict__ a, float* __restrict__ b,
                                      float invCount) {
  int c = threadIdx.x;
  float m   = sum[c] * invCount;
  float var = sq[c] * invCount - m * m;
  float av  = gamma[c] * rsqrtf(var + 1e-5f);
  a[c] = av;
  b[c] = beta[c] - m * av;
}

// out = base + a[col]*raw + b[col]   (residual + BN affine), 4 floats/thread
__global__ __launch_bounds__(256) void affine_residual_kernel(
    const float* __restrict__ base, const float* __restrict__ raw,
    const float* __restrict__ a, const float* __restrict__ b,
    float* __restrict__ out) {
  size_t i = ((size_t)blockIdx.x * 256 + threadIdx.x) * 4;
  int c = (int)(i & 127);
  f32x4 x  = *(const f32x4*)(base + i);
  f32x4 r  = *(const f32x4*)(raw + i);
  f32x4 a4 = *(const f32x4*)(a + c);
  f32x4 b4 = *(const f32x4*)(b + c);
  f32x4 o;
#pragma unroll
  for (int j = 0; j < 4; ++j) o[j] = x[j] + a4[j] * r[j] + b4[j];
  *(f32x4*)(out + i) = o;
}

// ---------------------------------------------------------------------------
extern "C" void kernel_launch(void* const* d_in, const int* in_sizes, int n_in,
                              void* d_out, int out_size, void* d_ws, size_t ws_size,
                              hipStream_t stream) {
  const float* node_fea = (const float*)d_in[0];
  const float* edge_fea = (const float*)d_in[1];
  const int*   idx1     = (const int*)d_in[2];
  const int*   idx2     = (const int*)d_in[3];
  const float* We1 = (const float*)d_in[4];
  const float* be1 = (const float*)d_in[5];
  const float* We2 = (const float*)d_in[6];
  const float* be2 = (const float*)d_in[7];
  const float* We3 = (const float*)d_in[8];
  const float* be3 = (const float*)d_in[9];
  const float* Wv1 = (const float*)d_in[10];
  const float* bv1 = (const float*)d_in[11];
  const float* Wv2 = (const float*)d_in[12];
  const float* bv2 = (const float*)d_in[13];
  const float* Wv3 = (const float*)d_in[14];
  const float* bv3 = (const float*)d_in[15];
  const float* gamma_e = (const float*)d_in[16];
  const float* beta_e  = (const float*)d_in[17];
  const float* gamma_v = (const float*)d_in[18];
  const float* beta_v  = (const float*)d_in[19];

  unsigned char* ws = (unsigned char*)d_ws;
  unsigned short* pw = (unsigned short*)ws;             // packed bf16 weights, 294912 B
  float* se_sum = (float*)(ws + 294912);                // 128
  float* se_sq  = se_sum + 128;
  float* sv_sum = se_sq + 128;
  float* sv_sq  = sv_sum + 128;
  float* a_e    = sv_sq + 128;
  float* b_e    = a_e + 128;
  float* a_v    = b_e + 128;
  float* b_v    = a_v + 128;
  float* cnt     = (float*)(ws + 294912 + 4096);        // 50000 f32
  float* seg_sum = (float*)(ws + 499200);               // 50000*128 f32 (25.6 MB)
  float* vi_raw  = (float*)(ws + 26099200);             // 50000*128 f32
  float* ek_raw  = (float*)(ws + 51699200);             // 640000*128 f32 (327.7 MB)

  float* out_node = (float*)d_out;                      // [NN,128] first
  float* out_edge = out_node + (size_t)NN * 128;        // then [NE,128]

  // Zero stats + cnt + seg_sum (graph-capture-safe async memset).
  hipMemsetAsync(ws + 294912, 0, (size_t)26099200 - 294912, stream);

  pack_weights_kernel<<<36, 256, 0, stream>>>(We1, We2, We3, Wv1, Wv2, Wv3, pw);

  edge_mlp_kernel<<<NE / 128, 256, 136192, stream>>>(
      node_fea, edge_fea, idx1, idx2,
      pw + 0, pw + 49152, pw + 65536, be1, be2, be3,
      ek_raw, seg_sum, cnt, se_sum, se_sq);

  finalize_stats_kernel<<<1, 128, 0, stream>>>(se_sum, se_sq, gamma_e, beta_e, a_e, b_e,
                                               1.0f / (float)NE);

  // edge_new = edge_fea + BN(ek_raw)
  affine_residual_kernel<<<(NE * 128) / 1024, 256, 0, stream>>>(edge_fea, ek_raw, a_e, b_e,
                                                                out_edge);

  node_mlp_kernel<<<(NN + 127) / 128, 256, 102400, stream>>>(
      node_fea, seg_sum, cnt, a_e, b_e,
      pw + 81920, pw + 114688, pw + 131072, bv1, bv2, bv3,
      vi_raw, sv_sum, sv_sq);

  finalize_stats_kernel<<<1, 128, 0, stream>>>(sv_sum, sv_sq, gamma_v, beta_v, a_v, b_v,
                                               1.0f / (float)NN);

  // node_new = node_fea + BN(vi_raw)
  affine_residual_kernel<<<(NN * 128) / 1024, 256, 0, stream>>>(node_fea, vi_raw, a_v, b_v,
                                                                out_node);
}
